// SA_Layer_4999341933051
// MI455X (gfx1250) — matrix operand order repaired and checked by reading, not compiled, not run on hardware
//
#include <hip/hip_runtime.h>
#include <hip/hip_bf16.h>

#define B_ 2
#define N_ 4096
#define D_ 64
#define P_ 64
#define H_ 256
#define K_ 16
#define BN_ (B_ * N_)   // 8192
#define EPS_ 1e-5f

typedef __attribute__((ext_vector_type(16))) _Float16 v16h;
typedef __attribute__((ext_vector_type(8)))  float    v8f;
typedef __attribute__((vector_size(16)))     int      v4i_cp;

// ---------------- async global->LDS copy (CDNA5 GLOBAL_LOAD_ASYNC_TO_LDS_B128) ----------------
#if __has_builtin(__builtin_amdgcn_global_load_async_to_lds_b128) && \
    __has_builtin(__builtin_amdgcn_s_wait_asynccnt)
#define ASYNC_CP 1
__device__ __forceinline__ void async_cp16(const void* g, void* l) {
  __builtin_amdgcn_global_load_async_to_lds_b128(
      (__attribute__((address_space(1))) v4i_cp*)g,
      (__attribute__((address_space(3))) v4i_cp*)l, 0, 0);
}
__device__ __forceinline__ void async_wait() { __builtin_amdgcn_s_wait_asynccnt(0); }
#else
#define ASYNC_CP 0
struct alignas(16) B16cp { unsigned int w[4]; };
__device__ __forceinline__ void async_cp16(const void* g, void* l) {
  *(B16cp*)l = *(const B16cp*)g;
}
__device__ __forceinline__ void async_wait() {}
#endif

// ---------------- WMMA fragment helpers (wave32, 16x16x32 f16) ----------------
// A-matrix 16x32 f16: lane L holds row M=L%16; element e -> K = e + 16*(L>>4)
__device__ __forceinline__ v16h load_A_frag(const _Float16* __restrict__ As, int lda,
                                            int kbase, int lane) {
  v16h a;
  const int m    = lane & 15;
  const int ksel = (lane >> 4) << 4;
  const _Float16* row = As + m * lda + kbase + ksel;
#pragma unroll
  for (int e = 0; e < 16; ++e) {
    int k = e;
    a[e] = row[k];
  }
  return a;
}

// B-matrix 32x16 f16 from an [N][Ktot] (pre-transposed) array:
// lane L holds column N=L%16; element e -> K = kbase + e + 16*(L>>4)  (32 contiguous bytes)
__device__ __forceinline__ v16h load_B_frag(const _Float16* __restrict__ Wt, int ldk,
                                            int n0, int kbase, int lane) {
  const _Float16* p = Wt + (size_t)(n0 + (lane & 15)) * ldk + kbase + ((lane >> 4) << 4);
  v16h b;
#pragma unroll
  for (int e = 0; e < 16; ++e) b[e] = p[e];
  return b;
}

__device__ __forceinline__ v8f wmma_f16(v16h a, v16h b, v8f c) {
  return __builtin_amdgcn_wmma_f32_16x16x32_f16(false, a, false, b, (short)0, c, false, false);
}

// ---------------- Kernel 0: convert weights to f16, [N][K] transposed ----------------
__global__ void prep_weights_kernel(const float* __restrict__ w_qkv, const float* __restrict__ w_p2,
                                    const float* __restrict__ w_a1,  const float* __restrict__ w_a2,
                                    const float* __restrict__ w_fc,
                                    _Float16* __restrict__ wqkvT, _Float16* __restrict__ wp2T,
                                    _Float16* __restrict__ wa1T,  _Float16* __restrict__ wa2T,
                                    _Float16* __restrict__ wfcT) {
  int i = blockIdx.x * blockDim.x + threadIdx.x;
  if (i < 192 * 64) { int n = i >> 6, k = i & 63;  wqkvT[i] = (_Float16)w_qkv[k * 192 + n]; }
  if (i < 64 * 64)  { int n = i >> 6, k = i & 63;  wp2T[i]  = (_Float16)w_p2[k * 64 + n]; }
  if (i < 256 * 64) { int n = i >> 6, k = i & 63;  wa1T[i]  = (_Float16)w_a1[k * 256 + n]; }
  if (i < 64 * 256) { int n = i >> 8, k = i & 255; wa2T[i]  = (_Float16)w_a2[k * 64 + n]; }
  if (i < 64 * 64)  { int n = i >> 6, k = i & 63;  wfcT[i]  = (_Float16)w_fc[k * 64 + n]; }
}

// ---------------- Kernel 1: KNN (top-16 by squared distance, LDS-tiled) ----------------
__global__ __launch_bounds__(256) void knn_kernel(const float* __restrict__ pos,
                                                  int* __restrict__ idx) {
  __shared__ float sp[256 * 3];
  const int tid = threadIdx.x;
  const int blk = blockIdx.x;          // b*16 + query_tile
  const int b   = blk >> 4;
  const int q   = ((blk & 15) << 8) + tid;
  const size_t qg = (size_t)b * N_ + q;
  const float qx = pos[qg * 3 + 0], qy = pos[qg * 3 + 1], qz = pos[qg * 3 + 2];

  float bd[16]; int bi[16];
#pragma unroll
  for (int t = 0; t < 16; ++t) { bd[t] = 3.4e38f; bi[t] = 0; }

  for (int nt = 0; nt < N_ / 256; ++nt) {
    const size_t cbase = (size_t)b * N_ + nt * 256;
    for (int j = tid; j < 256 * 3; j += 256) sp[j] = pos[cbase * 3 + j];
    __syncthreads();
#pragma unroll 4
    for (int j = 0; j < 256; ++j) {
      float dx = qx - sp[j * 3 + 0];
      float dy = qy - sp[j * 3 + 1];
      float dz = qz - sp[j * 3 + 2];
      float d2 = dx * dx + dy * dy + dz * dz;
      if (d2 < bd[15]) {
        bd[15] = d2; bi[15] = nt * 256 + j;
#pragma unroll
        for (int t = 15; t > 0; --t) {
          if (bd[t] < bd[t - 1]) {
            float td = bd[t]; bd[t] = bd[t - 1]; bd[t - 1] = td;
            int   ti = bi[t]; bi[t] = bi[t - 1]; bi[t - 1] = ti;
          }
        }
      }
    }
    __syncthreads();
  }
#pragma unroll
  for (int t = 0; t < 16; ++t) idx[qg * K_ + t] = bi[t];
}

// ---------------- Kernel 2: qkv = x @ w_qkv + b  (WMMA, 16-row tile per wave) ----------------
__global__ __launch_bounds__(32) void qkv_gemm_kernel(const float* __restrict__ x,
                                                      const _Float16* __restrict__ wqkvT,
                                                      const float* __restrict__ b_qkv,
                                                      float* __restrict__ qkv) {
  __shared__ float    Sx[16 * 64];
  __shared__ _Float16 As[16 * 64];
  const int lane = threadIdx.x;
  const size_t row0 = (size_t)blockIdx.x * 16;

  // async-stage the f32 A tile (4KB) into LDS, then convert to f16
  const char* gsrc = (const char*)(x + row0 * 64);
#pragma unroll
  for (int c = 0; c < 8; ++c)
    async_cp16(gsrc + (lane + 32 * c) * 16, (char*)Sx + (lane + 32 * c) * 16);
  async_wait();
  __syncthreads();
  for (int i = lane; i < 16 * 64; i += 32) As[i] = (_Float16)Sx[i];
  __syncthreads();

  v16h a0 = load_A_frag(As, 64, 0, lane);
  v16h a1 = load_A_frag(As, 64, 32, lane);
  const int colb = lane & 15;
  const int mhi  = (lane >> 4) << 3;
#pragma unroll
  for (int t = 0; t < 12; ++t) {
    v8f c = {};
    c = wmma_f16(a0, load_B_frag(wqkvT, 64, t * 16, 0, lane), c);
    c = wmma_f16(a1, load_B_frag(wqkvT, 64, t * 16, 32, lane), c);
    int col = t * 16 + colb;
    float bias = b_qkv[col];
#pragma unroll
    for (int r = 0; r < 8; ++r)
      qkv[(row0 + r + mhi) * 192 + col] = c[r] + bias;
  }
}

// ---------------- Kernel 3: zero BN stats ----------------
__global__ void zero_stats_kernel(float* __restrict__ stats) {
  if (threadIdx.x < 128) stats[threadIdx.x] = 0.f;
}

// ---------------- Kernel 4: fused per-query attention (1 wave = 1 query, M=16=K nbrs) ----------------
__global__ __launch_bounds__(32) void attn_kernel(const float* __restrict__ qkv,
                                                  const int* __restrict__ idx,
                                                  const float* __restrict__ pos,
                                                  const float* __restrict__ w_p1,
                                                  const float* __restrict__ b_p1,
                                                  const _Float16* __restrict__ wp2T,
                                                  const float* __restrict__ b_p2,
                                                  const _Float16* __restrict__ wa1T,
                                                  const float* __restrict__ b_a1,
                                                  const _Float16* __restrict__ wa2T,
                                                  const float* __restrict__ b_a2,
                                                  float* __restrict__ agg) {
  __shared__ _Float16 sPH[16 * 64];    // relu(rel @ w_p1 + b_p1), f16
  __shared__ _Float16 sH[16 * 64];     // h = qk_map + pos_emb, f16
  __shared__ _Float16 sHID[16 * 256];  // relu(h @ w_a1 + b_a1), f16
  __shared__ float    sKV[16 * 128];   // per neighbor row: [0..63]=k, [64..127]=v
  __shared__ float    sQ[64];
  __shared__ int      sIdx[16];

  const int lane = threadIdx.x;
  const int bn   = blockIdx.x;       // 0..B*N-1
  const int b    = bn / N_;

  if (lane < 16) sIdx[lane] = idx[(size_t)bn * K_ + lane];
  sQ[lane]      = qkv[(size_t)bn * 192 + lane];
  sQ[lane + 32] = qkv[(size_t)bn * 192 + lane + 32];
  __syncthreads();

  // async gather of neighbor k|v rows (contiguous 128 floats per row) into LDS
  // 16 rows x 32 chunks of 16B = 512 chunks, 16 per lane
  for (int c = lane; c < 512; c += 32) {
    const int r = c >> 5, part = c & 31;
    const size_t ng = (size_t)b * N_ + sIdx[r];
    async_cp16(qkv + ng * 192 + 64 + part * 4, &sKV[r * 128 + part * 4]);
  }

  const float px = pos[(size_t)bn * 3 + 0];
  const float py = pos[(size_t)bn * 3 + 1];
  const float pz = pos[(size_t)bn * 3 + 2];

  // pos-MLP first layer (tiny, VALU) overlapped with the async gather
  for (int r = 0; r < 16; ++r) {
    const size_t ng = (size_t)b * N_ + sIdx[r];
    const float rx = px - pos[ng * 3 + 0];
    const float ry = py - pos[ng * 3 + 1];
    const float rz = pz - pos[ng * 3 + 2];
    for (int c = lane; c < 64; c += 32) {
      float h = rx * w_p1[c] + ry * w_p1[64 + c] + rz * w_p1[128 + c] + b_p1[c];
      sPH[r * 64 + c] = (_Float16)fmaxf(h, 0.f);
    }
  }
  async_wait();
  __syncthreads();

  const int colb = lane & 15;
  const int mhi  = (lane >> 4) << 3;

  // pos_emb = sPH @ w_p2 + b_p2 ; h = (q - k_nb) + pos_emb     (8 WMMAs)
  {
    v16h a0 = load_A_frag(sPH, 64, 0, lane);
    v16h a1 = load_A_frag(sPH, 64, 32, lane);
#pragma unroll
    for (int t = 0; t < 4; ++t) {
      v8f c = {};
      c = wmma_f16(a0, load_B_frag(wp2T, 64, t * 16, 0, lane), c);
      c = wmma_f16(a1, load_B_frag(wp2T, 64, t * 16, 32, lane), c);
      int col = t * 16 + colb;
#pragma unroll
      for (int r = 0; r < 8; ++r) {
        int m = r + mhi;
        float h = c[r] + b_p2[col] + sQ[col] - sKV[m * 128 + col];
        sH[m * 64 + col] = (_Float16)h;
      }
    }
  }
  __syncthreads();

  // layer1: hid = relu(h @ w_a1 + b_a1)   [16,64]x[64,256]   (32 WMMAs)
  {
    v16h a0 = load_A_frag(sH, 64, 0, lane);
    v16h a1 = load_A_frag(sH, 64, 32, lane);
#pragma unroll
    for (int t = 0; t < 16; ++t) {
      v8f c = {};
      c = wmma_f16(a0, load_B_frag(wa1T, 64, t * 16, 0, lane), c);
      c = wmma_f16(a1, load_B_frag(wa1T, 64, t * 16, 32, lane), c);
      int col = t * 16 + colb;
#pragma unroll
      for (int r = 0; r < 8; ++r) {
        int m = r + mhi;
        sHID[m * 256 + col] = (_Float16)fmaxf(c[r] + b_a1[col], 0.f);
      }
    }
  }
  __syncthreads();

  // layer2 + softmax over K (rows) + weighted agg   [16,256]x[256,64]  (32 WMMAs)
#pragma unroll
  for (int t = 0; t < 4; ++t) {
    v8f c = {};
#pragma unroll
    for (int kk = 0; kk < 8; ++kk) {
      v16h a  = load_A_frag(sHID, 256, kk * 32, lane);
      v16h bf = load_B_frag(wa2T, 256, t * 16, kk * 32, lane);
      c = wmma_f16(a, bf, c);
    }
    int col = t * 16 + colb;
    float s[8];
    float mx = -3.0e38f;
#pragma unroll
    for (int r = 0; r < 8; ++r) { s[r] = c[r] + b_a2[col]; mx = fmaxf(mx, s[r]); }
    mx = fmaxf(mx, __shfl_xor(mx, 16));          // column max over all 16 rows
    float sum = 0.f, acc = 0.f;
#pragma unroll
    for (int r = 0; r < 8; ++r) {
      float e = __expf(s[r] - mx);
      sum += e;
      acc += e * sKV[(r + mhi) * 128 + 64 + col];
    }
    sum += __shfl_xor(sum, 16);
    acc += __shfl_xor(acc, 16);
    if (lane < 16) agg[(size_t)bn * 64 + col] = acc / sum;
  }
}

// ---------------- Kernel 5: fc GEMM + BN statistics (atomics) ----------------
__global__ __launch_bounds__(32) void fc_gemm_kernel(const float* __restrict__ aggin,
                                                     const _Float16* __restrict__ wfcT,
                                                     const float* __restrict__ b_fc,
                                                     float* __restrict__ fco,
                                                     float* __restrict__ stats) {
  __shared__ float    Sx[16 * 64];
  __shared__ _Float16 As[16 * 64];
  const int lane = threadIdx.x;
  const size_t row0 = (size_t)blockIdx.x * 16;

  const char* gsrc = (const char*)(aggin + row0 * 64);
#pragma unroll
  for (int c = 0; c < 8; ++c)
    async_cp16(gsrc + (lane + 32 * c) * 16, (char*)Sx + (lane + 32 * c) * 16);
  async_wait();
  __syncthreads();
  for (int i = lane; i < 16 * 64; i += 32) As[i] = (_Float16)Sx[i];
  __syncthreads();

  v16h a0 = load_A_frag(As, 64, 0, lane);
  v16h a1 = load_A_frag(As, 64, 32, lane);
  const int colb = lane & 15;
  const int mhi  = (lane >> 4) << 3;
#pragma unroll
  for (int t = 0; t < 4; ++t) {
    v8f c = {};
    c = wmma_f16(a0, load_B_frag(wfcT, 64, t * 16, 0, lane), c);
    c = wmma_f16(a1, load_B_frag(wfcT, 64, t * 16, 32, lane), c);
    int col = t * 16 + colb;
    float bias = b_fc[col];
    float s1 = 0.f, s2 = 0.f;
#pragma unroll
    for (int r = 0; r < 8; ++r) {
      float v = c[r] + bias;
      fco[(row0 + r + mhi) * 64 + col] = v;
      s1 += v; s2 += v * v;
    }
    s1 += __shfl_xor(s1, 16);
    s2 += __shfl_xor(s2, 16);
    if (lane < 16) {
      atomicAdd(&stats[col], s1);
      atomicAdd(&stats[64 + col], s2);
    }
  }
}

// ---------------- Kernel 6: BN normalize + ReLU + residual ----------------
__global__ void bn_relu_res_kernel(const float* __restrict__ fco, const float* __restrict__ stats,
                                   const float* __restrict__ gamma, const float* __restrict__ beta,
                                   const float* __restrict__ x, float* __restrict__ out, int total) {
  int i = blockIdx.x * blockDim.x + threadIdx.x;
  if (i >= total) return;
  int d = i & 63;
  const float inv = 1.0f / (float)BN_;
  float mean = stats[d] * inv;
  float var  = stats[64 + d] * inv - mean * mean;
  float v = (fco[i] - mean) * rsqrtf(var + EPS_) * gamma[d] + beta[d];
  out[i] = fmaxf(v, 0.f) + x[i];
}

// ---------------- Launcher ----------------
extern "C" void kernel_launch(void* const* d_in, const int* in_sizes, int n_in,
                              void* d_out, int out_size, void* d_ws, size_t ws_size,
                              hipStream_t stream) {
  const float* x     = (const float*)d_in[0];
  const float* pos   = (const float*)d_in[1];
  const float* w_qkv = (const float*)d_in[2];
  const float* b_qkv = (const float*)d_in[3];
  const float* w_p1  = (const float*)d_in[4];
  const float* b_p1  = (const float*)d_in[5];
  const float* w_p2  = (const float*)d_in[6];
  const float* b_p2  = (const float*)d_in[7];
  const float* w_a1  = (const float*)d_in[8];
  const float* b_a1  = (const float*)d_in[9];
  const float* w_a2  = (const float*)d_in[10];
  const float* b_a2  = (const float*)d_in[11];
  const float* w_fc  = (const float*)d_in[12];
  const float* b_fc  = (const float*)d_in[13];
  const float* gamma = (const float*)d_in[14];
  const float* beta  = (const float*)d_in[15];
  float* out = (float*)d_out;

  // workspace layout (256B aligned chunks)
  char* base = (char*)d_ws;
  size_t off = 0;
  auto alloc = [&](size_t bytes) { void* p = base + off; off = (off + bytes + 255) & ~(size_t)255; return p; };
  float*    qkv   = (float*)alloc((size_t)BN_ * 192 * 4);
  float*    agg   = (float*)alloc((size_t)BN_ * 64 * 4);
  float*    fco   = (float*)alloc((size_t)BN_ * 64 * 4);
  int*      idx   = (int*)alloc((size_t)BN_ * K_ * 4);
  float*    stats = (float*)alloc(128 * 4);
  _Float16* wqkvT = (_Float16*)alloc(192 * 64 * 2);
  _Float16* wp2T  = (_Float16*)alloc(64 * 64 * 2);
  _Float16* wa1T  = (_Float16*)alloc(256 * 64 * 2);
  _Float16* wa2T  = (_Float16*)alloc(64 * 256 * 2);
  _Float16* wfcT  = (_Float16*)alloc(64 * 64 * 2);

  prep_weights_kernel<<<64, 256, 0, stream>>>(w_qkv, w_p2, w_a1, w_a2, w_fc,
                                              wqkvT, wp2T, wa1T, wa2T, wfcT);
  knn_kernel<<<B_ * (N_ / 256), 256, 0, stream>>>(pos, idx);
  qkv_gemm_kernel<<<BN_ / 16, 32, 0, stream>>>(x, wqkvT, b_qkv, qkv);
  zero_stats_kernel<<<1, 128, 0, stream>>>(stats);
  attn_kernel<<<BN_, 32, 0, stream>>>(qkv, idx, pos, w_p1, b_p1, wp2T, b_p2,
                                      wa1T, b_a1, wa2T, b_a2, agg);
  fc_gemm_kernel<<<BN_ / 16, 32, 0, stream>>>(agg, wfcT, b_fc, fco, stats);
  bn_relu_res_kernel<<<(BN_ * 64 + 255) / 256, 256, 0, stream>>>(fco, stats, gamma, beta, x,
                                                                 out, BN_ * 64);
}
